// GNN_30940944401187
// MI455X (gfx1250) — compile-verified, matrix operand order checked by reading, not run on hardware
//
#include <hip/hip_runtime.h>

typedef __attribute__((ext_vector_type(16))) _Float16 v16h;
typedef __attribute__((ext_vector_type(8)))  float    v8f;

#define HDIM 64

// ---------------- utility ----------------
__global__ void fill0_kernel(float* p, int n) {
    int i = blockIdx.x * blockDim.x + threadIdx.x;
    if (i < n) p[i] = 0.0f;
}

// Pack [W_rel(64x64); W_root(64x64)] -> WMMA B-fragment order, f16.
// Fragment index fidx = (kc*4 + t)*32 + lane, element j in [0,16):
//   col = t*16 + (lane&15), half = lane>>4
//   kk  = (j<8) ? half*8 + j : 16 + half*8 + (j-8);  k = kc*32 + kk
__global__ void build_wfrag_kernel(const float* __restrict__ Wrel,
                                   const float* __restrict__ Wroot,
                                   _Float16* __restrict__ wfrag) {
    int tid = blockIdx.x * blockDim.x + threadIdx.x;   // 4*4*32*16 = 8192
    if (tid >= 8192) return;
    int j    = tid & 15;
    int lane = (tid >> 4) & 31;
    int t    = (tid >> 9) & 3;
    int kc   = tid >> 11;
    int half = lane >> 4;
    int col  = t * 16 + (lane & 15);
    int kk   = (j < 8) ? (half * 8 + j) : (16 + half * 8 + (j - 8));
    int k    = kc * 32 + kk;
    float v  = (k < 64) ? Wrel[k * HDIM + col] : Wroot[(k - 64) * HDIM + col];
    wfrag[tid] = (_Float16)v;
}

// ---------------- layer 1 (C_IN = 1, rank-1) ----------------
__global__ void l1_scatter_kernel(const float* __restrict__ x, const int* __restrict__ ei,
                                  const float* __restrict__ ew, float* __restrict__ agg1, int E) {
    int e = blockIdx.x * blockDim.x + threadIdx.x;
    if (e >= E) return;
    int s = ei[e];
    int d = ei[E + e];
    atomicAdd(&agg1[d], x[s] * ew[e]);
}

__global__ void l1_node_kernel(const float* __restrict__ x, const float* __restrict__ agg1,
                               const float* __restrict__ W1rel, const float* __restrict__ b1,
                               const float* __restrict__ W1root, float* __restrict__ h1, int N) {
    int tid = blockIdx.x * blockDim.x + threadIdx.x;
    if (tid >= N * HDIM) return;
    int n = tid >> 6, h = tid & 63;
    float v = agg1[n] * W1rel[h] + b1[h] + x[n] * W1root[h];
    h1[tid] = fmaxf(v, 0.0f);
}

// ---------------- 64-wide edge scatter (message + segment_sum) ----------------
__global__ void scatter64_kernel(const float* __restrict__ hsrc, const int* __restrict__ ei,
                                 const float* __restrict__ ew, float* __restrict__ agg, int E) {
    int tid = blockIdx.x * blockDim.x + threadIdx.x;
    if (tid >= E * 16) return;
    int e = tid >> 4;
    int f = (tid & 15) << 2;
    int s = ei[e];
    int d = ei[E + e];
    if ((tid & 15) == 0 && (e + 64) < E) {
        // hint the upcoming gather row toward the caches (global_prefetch_b8)
        __builtin_prefetch(&hsrc[(size_t)ei[e + 64] * HDIM], 0, 1);
    }
    float w = ew[e];
    const float4 hv = *reinterpret_cast<const float4*>(&hsrc[(size_t)s * HDIM + f]);
    float* ap = &agg[(size_t)d * HDIM + f];
    atomicAdd(ap + 0, hv.x * w);
    atomicAdd(ap + 1, hv.y * w);
    atomicAdd(ap + 2, hv.z * w);
    atomicAdd(ap + 3, hv.w * w);
}

// ---------------- fused GEMM: out = [agg | h] @ [W_rel; W_root] + b, optional ReLU ----------------
// Block = 128 threads = 4 wave32 waves; block covers 64 rows; wave covers 16 rows.
// K = 128 (agg: 0..63, h: 64..127), 4 k-chunks of 32; 4 column tiles of 16.
// Per k-chunk: batch all 8 ds_load_b128 (4 B fragments) before the 4 WMMAs so a
// single dscnt wait covers the chunk instead of one wait per WMMA.
__global__ __launch_bounds__(128)
void gemm_wmma_kernel(const float* __restrict__ hprev, const float* __restrict__ agg,
                      const _Float16* __restrict__ wfrag, const float* __restrict__ bias,
                      float* __restrict__ out, int N, int do_relu) {
    __shared__ __align__(32) _Float16 wlds[8192];     // 16 KB: full packed weight block
    {
        const uint4* src = reinterpret_cast<const uint4*>(wfrag);
        uint4* dst = reinterpret_cast<uint4*>(wlds);
        for (int i = threadIdx.x; i < 1024; i += 128) dst[i] = src[i];
    }
    __syncthreads();

    const int wave = threadIdx.x >> 5;
    const int lane = threadIdx.x & 31;
    const int rowbase = blockIdx.x * 64 + wave * 16;
    const int m = lane & 15;
    const int half = lane >> 4;

    int r = rowbase + m;
    if (r >= N) r = N - 1;                            // clamp OOB lanes (stores guarded)
    const float* aggrow = agg   + (size_t)r * HDIM;
    const float* hrow   = hprev + (size_t)r * HDIM;

    v8f acc[4] = {v8f{}, v8f{}, v8f{}, v8f{}};
    const v16h* wv = reinterpret_cast<const v16h*>(wlds);

#pragma unroll
    for (int kc = 0; kc < 4; ++kc) {
        const float* srcrow = (kc < 2) ? aggrow : hrow;
        const int k0 = (kc & 1) * 32;

        // Batch-load the 4 B fragments for this k-chunk from LDS first.
        v16h bfrag[4];
#pragma unroll
        for (int t = 0; t < 4; ++t) bfrag[t] = wv[(kc * 4 + t) * 32 + lane];

        // A fragment: lane m holds K = {half*8 .. half*8+7} and {16+half*8 .. 16+half*8+7}
        const float4* p0 = reinterpret_cast<const float4*>(srcrow + k0 + half * 8);
        const float4* p1 = reinterpret_cast<const float4*>(srcrow + k0 + 16 + half * 8);
        float4 a0 = p0[0], a1 = p0[1];
        float4 c0 = p1[0], c1 = p1[1];
        v16h afrag;
        afrag[0]  = (_Float16)a0.x; afrag[1]  = (_Float16)a0.y;
        afrag[2]  = (_Float16)a0.z; afrag[3]  = (_Float16)a0.w;
        afrag[4]  = (_Float16)a1.x; afrag[5]  = (_Float16)a1.y;
        afrag[6]  = (_Float16)a1.z; afrag[7]  = (_Float16)a1.w;
        afrag[8]  = (_Float16)c0.x; afrag[9]  = (_Float16)c0.y;
        afrag[10] = (_Float16)c0.z; afrag[11] = (_Float16)c0.w;
        afrag[12] = (_Float16)c1.x; afrag[13] = (_Float16)c1.y;
        afrag[14] = (_Float16)c1.z; afrag[15] = (_Float16)c1.w;

#pragma unroll
        for (int t = 0; t < 4; ++t) {
            acc[t] = __builtin_amdgcn_wmma_f32_16x16x32_f16(
                false, afrag, false, bfrag[t], (short)0, acc[t], false, false);
        }
    }

    // Epilogue: C layout — col = lane&15, row = rowbase + j + (lane>=16)*8
    const int col0 = lane & 15;
#pragma unroll
    for (int t = 0; t < 4; ++t) {
        const int col = t * 16 + col0;
        const float bc = bias[col];
#pragma unroll
        for (int j = 0; j < 8; ++j) {
            int row = rowbase + j + half * 8;
            if (row < N) {
                float v = acc[t][j] + bc;
                if (do_relu) v = fmaxf(v, 0.0f);
                out[(size_t)row * HDIM + col] = v;
            }
        }
    }
}

// ---------------- global mean pool + head ----------------
__global__ void pool_scatter_kernel(const float* __restrict__ h3, const int* __restrict__ batch,
                                    float* __restrict__ sums, int N) {
    int tid = blockIdx.x * blockDim.x + threadIdx.x;
    if (tid >= N * 16) return;
    int n = tid >> 4;
    int f = (tid & 15) << 2;
    const float4 hv = *reinterpret_cast<const float4*>(&h3[(size_t)n * HDIM + f]);
    float* sp = &sums[(size_t)batch[n] * HDIM + f];
    atomicAdd(sp + 0, hv.x);
    atomicAdd(sp + 1, hv.y);
    atomicAdd(sp + 2, hv.z);
    atomicAdd(sp + 3, hv.w);
}

__global__ void pool_count_kernel(const int* __restrict__ batch, float* __restrict__ cnt, int N) {
    int n = blockIdx.x * blockDim.x + threadIdx.x;
    if (n < N) atomicAdd(&cnt[batch[n]], 1.0f);
}

__global__ void head_kernel(const float* __restrict__ sums, const float* __restrict__ cnt,
                            const float* __restrict__ Wlin, const float* __restrict__ blin,
                            float* __restrict__ out, int G) {
    int tid = blockIdx.x * blockDim.x + threadIdx.x;
    if (tid >= G * 2) return;
    int g = tid >> 1, c = tid & 1;
    float inv = 1.0f / fmaxf(cnt[g], 1.0f);
    float acc = blin[c];
#pragma unroll 8
    for (int k = 0; k < HDIM; ++k) acc += sums[g * HDIM + k] * inv * Wlin[k * 2 + c];
    out[tid] = acc;
}

// ---------------- launch ----------------
extern "C" void kernel_launch(void* const* d_in, const int* in_sizes, int n_in,
                              void* d_out, int out_size, void* d_ws, size_t ws_size,
                              hipStream_t stream) {
    const float* x      = (const float*)d_in[0];
    const int*   ei     = (const int*)  d_in[1];
    const int*   batch  = (const int*)  d_in[2];
    const float* ew     = (const float*)d_in[3];
    const float* W1rel  = (const float*)d_in[4];
    const float* b1     = (const float*)d_in[5];
    const float* W1root = (const float*)d_in[6];
    const float* W2rel  = (const float*)d_in[7];
    const float* b2     = (const float*)d_in[8];
    const float* W2root = (const float*)d_in[9];
    const float* W3rel  = (const float*)d_in[10];
    const float* b3     = (const float*)d_in[11];
    const float* W3root = (const float*)d_in[12];
    const float* Wlin   = (const float*)d_in[13];
    const float* blin   = (const float*)d_in[14];
    float* out = (float*)d_out;

    const int N = in_sizes[0];      // x is [N,1]
    const int E = in_sizes[3];      // edge_weight is [E]
    const int G = out_size / 2;

    char* ws = (char*)d_ws;
    size_t off = 0;
    auto alloc = [&](size_t bytes) {
        void* p = ws + off;
        off = (off + bytes + 255) & ~(size_t)255;
        return p;
    };
    float*     bufA   = (float*)    alloc((size_t)N * HDIM * 4);
    float*     bufB   = (float*)    alloc((size_t)N * HDIM * 4);
    float*     aggbuf = (float*)    alloc((size_t)N * HDIM * 4);
    float*     agg1   = (float*)    alloc((size_t)N * 4);
    _Float16*  wfrag2 = (_Float16*) alloc(8192 * 2);
    _Float16*  wfrag3 = (_Float16*) alloc(8192 * 2);
    float*     sums   = (float*)    alloc((size_t)G * HDIM * 4);
    float*     cnt    = (float*)    alloc((size_t)G * 4);

    const int T = 256;
    const int gemm_blocks = (N + 63) / 64;

    // pack weights into WMMA fragment order (f16)
    build_wfrag_kernel<<<(8192 + T - 1) / T, T, 0, stream>>>(W2rel, W2root, wfrag2);
    build_wfrag_kernel<<<(8192 + T - 1) / T, T, 0, stream>>>(W3rel, W3root, wfrag3);

    // ---- layer 1 (rank-1) ----
    fill0_kernel<<<(N + T - 1) / T, T, 0, stream>>>(agg1, N);
    l1_scatter_kernel<<<(E + T - 1) / T, T, 0, stream>>>(x, ei, ew, agg1, E);
    l1_node_kernel<<<((N * HDIM) + T - 1) / T, T, 0, stream>>>(x, agg1, W1rel, b1, W1root, bufA, N);

    // ---- layer 2 ----
    fill0_kernel<<<((N * HDIM) + T - 1) / T, T, 0, stream>>>(aggbuf, N * HDIM);
    scatter64_kernel<<<((E * 16) + T - 1) / T, T, 0, stream>>>(bufA, ei, ew, aggbuf, E);
    gemm_wmma_kernel<<<gemm_blocks, 128, 0, stream>>>(bufA, aggbuf, wfrag2, b2, bufB, N, 1);

    // ---- layer 3 ----
    fill0_kernel<<<((N * HDIM) + T - 1) / T, T, 0, stream>>>(aggbuf, N * HDIM);
    scatter64_kernel<<<((E * 16) + T - 1) / T, T, 0, stream>>>(bufB, ei, ew, aggbuf, E);
    gemm_wmma_kernel<<<gemm_blocks, 128, 0, stream>>>(bufB, aggbuf, wfrag3, b3, bufA, N, 0);

    // ---- mean pool + head ----
    fill0_kernel<<<((G * HDIM) + T - 1) / T, T, 0, stream>>>(sums, G * HDIM);
    fill0_kernel<<<(G + T - 1) / T, T, 0, stream>>>(cnt, G);
    pool_scatter_kernel<<<((N * 16) + T - 1) / T, T, 0, stream>>>(bufA, batch, sums, N);
    pool_count_kernel<<<(N + T - 1) / T, T, 0, stream>>>(batch, cnt, N);
    head_kernel<<<((G * 2) + T - 1) / T, T, 0, stream>>>(sums, cnt, Wlin, blin, out, G);
}
